// MoELayer_8065948582286
// MI455X (gfx1250) — matrix operand order repaired and checked by reading, not compiled, not run on hardware
//
#include <hip/hip_runtime.h>

// ---------------------------------------------------------------------------
// MoE layer for MI455X (gfx1250): bf16 WMMA (V_WMMA_F32_16X16X32_BF16) path.
// N=8192 tokens, D=2048, H=1408, E=8 top-2 routed experts + 1 shared expert.
// Compute-bound (~1.28 dense TFLOP vs ~380MB HBM traffic); bf16 weights
// (~156MB) fit in the 192MB L2. TM=32 token tiles (222KB LDS of CDNA5's
// 320KB/WGP) so every B fragment feeds two 16x16x32 WMMAs.
// ---------------------------------------------------------------------------

#define D_DIM 2048
#define H_DIM 1408
#define E_NUM 8
#define N_TOK 8192
#define TM    32            // tokens per workgroup tile (two 16-row M tiles)
#define SX    2056          // LDS row stride for x tile  (2056*2B/4 % 64 = 4 -> conflict-free)
#define SH    1416          // LDS row stride for hidden  (1416*2B/4 % 64 = 4)

typedef __attribute__((ext_vector_type(16))) __bf16 v16bf;
typedef __attribute__((ext_vector_type(8)))  float  v8f;

union BPack { v16bf v; uint4 q[2]; };

static __device__ __forceinline__ v16bf pack2(uint4 lo, uint4 hi) {
    BPack p; p.q[0] = lo; p.q[1] = hi; return p.v;
}

// round-to-nearest-even float -> bf16 bits
static __device__ __forceinline__ unsigned short f2bf(float f) {
    unsigned int u = __float_as_uint(f);
    unsigned int r = u + 0x7FFFu + ((u >> 16) & 1u);
    return (unsigned short)(r >> 16);
}

// silu with fast v_rcp_f32 (bf16-bound data; full IEEE divide is wasted VALU)
static __device__ __forceinline__ float silu_fast(float g) {
    return g * __builtin_amdgcn_rcpf(1.0f + __expf(-g));
}

// ---------------------------------------------------------------------------
// Stage 1: fp32 -> bf16 bulk conversion (grid-stride)
// ---------------------------------------------------------------------------
__global__ void cvt_bf16_kernel(const float* __restrict__ in,
                                unsigned short* __restrict__ out, int n) {
    for (int i = blockIdx.x * blockDim.x + threadIdx.x; i < n;
         i += gridDim.x * blockDim.x)
        out[i] = f2bf(in[i]);
}

// ---------------------------------------------------------------------------
// Stage 2: gating. One wave32 per token: logits[8] -> softmax, top-2 on
// biased logits, normalized combine weights into cw[N, E].
// ---------------------------------------------------------------------------
__global__ __launch_bounds__(256) void gate_kernel(const float* __restrict__ x,
                                                   const float* __restrict__ gw,
                                                   const float* __restrict__ bias,
                                                   float* __restrict__ cw) {
    const int lane = threadIdx.x & 31;
    const int wave = threadIdx.x >> 5;
    const int tok  = blockIdx.x * 8 + wave;
    if (tok >= N_TOK) return;

    float acc[E_NUM];
#pragma unroll
    for (int e = 0; e < E_NUM; ++e) acc[e] = 0.0f;

    const float* xr = x + (size_t)tok * D_DIM;
    for (int d = lane; d < D_DIM; d += 32) {
        float xv = xr[d];
#pragma unroll
        for (int e = 0; e < E_NUM; ++e) acc[e] += xv * gw[e * D_DIM + d];
    }
#pragma unroll
    for (int e = 0; e < E_NUM; ++e)
        for (int off = 16; off > 0; off >>= 1)
            acc[e] += __shfl_down(acc[e], off, 32);

    if (lane == 0) {
        float mx = acc[0];
#pragma unroll
        for (int e = 1; e < E_NUM; ++e) mx = fmaxf(mx, acc[e]);
        float p[E_NUM]; float den = 0.0f;
#pragma unroll
        for (int e = 0; e < E_NUM; ++e) { p[e] = __expf(acc[e] - mx); den += p[e]; }
        float rden = __builtin_amdgcn_rcpf(den);
#pragma unroll
        for (int e = 0; e < E_NUM; ++e) p[e] *= rden;

        float b[E_NUM];
#pragma unroll
        for (int e = 0; e < E_NUM; ++e) b[e] = acc[e] + bias[e];
        int i1 = 0;
#pragma unroll
        for (int e = 1; e < E_NUM; ++e) if (b[e] > b[i1]) i1 = e;
        int i2 = (i1 == 0) ? 1 : 0;
#pragma unroll
        for (int e = 0; e < E_NUM; ++e)
            if (e != i1 && b[e] > b[i2]) i2 = e;

        float rs = __builtin_amdgcn_rcpf(p[i1] + p[i2]);
#pragma unroll
        for (int e = 0; e < E_NUM; ++e) cw[(size_t)tok * E_NUM + e] = 0.0f;
        cw[(size_t)tok * E_NUM + i1] = p[i1] * rs;
        cw[(size_t)tok * E_NUM + i2] = p[i2] * rs;
    }
}

// ---------------------------------------------------------------------------
// Stage 3: fused expert compute. One WG (8 waves) per 32-token tile.
// Loops 9 "experts" (slot 0 = shared, weight 1.0; slots 1..8 = routed,
// skipped if no token in the tile routes to them).
//   Phase A: hidden[32,1408] = silu(x Wg^T) * (x Wu^T) * w[token]  (bf16, LDS)
//   Phase B: out[32,2048]   += hidden Wd^T                          (f32, global)
// WMMA fragment layouts per CDNA5 ISA:
//   A (16-bit 16x32): lane<16 -> M=lane, K chunks {k..k+7, k+16..k+23};
//                     lane>=16 -> same M, K chunks {k+8.., k+24..}  (2x b128)
//   B (16-bit 32x16): lane<16 -> N=lane, K chunks {k..k+7, k+16..k+23}; lane>=16 -> {k+8.., k+24..}
//                     == same K order as A, from weight row N (2x b128 from global)
//   C (f32 16x16): VGPR v, lane l -> (M = v + 8*(l>>4), N = l&15)
// ---------------------------------------------------------------------------
__global__ __launch_bounds__(256) void moe_kernel(
    const unsigned short* __restrict__ xb,
    const unsigned short* __restrict__ Wgb,
    const unsigned short* __restrict__ Wub,
    const unsigned short* __restrict__ Wdb,
    const unsigned short* __restrict__ Wsg,
    const unsigned short* __restrict__ Wsu,
    const unsigned short* __restrict__ Wsd,
    const float* __restrict__ cw,
    float* __restrict__ out) {

    __shared__ unsigned short xs[TM * SX];   // 131,584 B
    __shared__ unsigned short hs[TM * SH];   //  90,624 B
    __shared__ float wtile[TM];
    __shared__ int active;

    const int tid  = threadIdx.x;
    const int lane = tid & 31;
    const int wave = tid >> 5;
    const int hgrp = lane >> 4;    // K-half group within the wave
    const int r15  = lane & 15;    // A-row / B-column within the 16x16 tile
    const int tok0 = blockIdx.x * TM;

    // Stage the 32 x 2048 bf16 x-tile into LDS (b128 chunks).
    for (int c = tid; c < TM * (D_DIM / 8); c += 256) {
        int row = c >> 8;              // D/8 == 256
        int kc  = (c & 255) * 8;
        *(uint4*)(xs + row * SX + kc) =
            *(const uint4*)(xb + (size_t)(tok0 + row) * D_DIM + kc);
    }
    __syncthreads();

    for (int ee = 0; ee < 9; ++ee) {
        if (tid == 0) active = 0;
        __syncthreads();
        if (tid < TM) {
            float w = (ee == 0) ? 1.0f
                                : cw[(size_t)(tok0 + tid) * E_NUM + (ee - 1)];
            wtile[tid] = w;
            if (w != 0.0f) atomicOr(&active, 1);
        }
        __syncthreads();
        int act = active;
        __syncthreads();
        if (!act) continue;

        const unsigned short* Wg_e =
            ee ? Wgb + (size_t)(ee - 1) * H_DIM * D_DIM : Wsg;
        const unsigned short* Wu_e =
            ee ? Wub + (size_t)(ee - 1) * H_DIM * D_DIM : Wsu;
        const unsigned short* Wd_e =
            ee ? Wdb + (size_t)(ee - 1) * D_DIM * H_DIM : Wsd;

        // ----- Phase A: hidden = silu(x Wg^T) * (x Wu^T) * w -----
        for (int nc = 0; nc < H_DIM; nc += 128) {     // 8 waves x 16 cols
            const int n0 = nc + wave * 16;
            v8f ag0 = {0.f,0.f,0.f,0.f,0.f,0.f,0.f,0.f};
            v8f ag1 = {0.f,0.f,0.f,0.f,0.f,0.f,0.f,0.f};
            v8f au0 = {0.f,0.f,0.f,0.f,0.f,0.f,0.f,0.f};
            v8f au1 = {0.f,0.f,0.f,0.f,0.f,0.f,0.f,0.f};
            const unsigned short* ap0 = xs + r15 * SX + hgrp * 8;
            const unsigned short* ap1 = xs + (r15 + 16) * SX + hgrp * 8;
            const unsigned short* bg = Wg_e + (size_t)(n0 + r15) * D_DIM + hgrp * 8;
            const unsigned short* bu = Wu_e + (size_t)(n0 + r15) * D_DIM + hgrp * 8;
#pragma unroll 2
            for (int k = 0; k < D_DIM; k += 32) {
                v16bf a0 = pack2(*(const uint4*)(ap0 + k),
                                 *(const uint4*)(ap0 + k + 16));
                v16bf a1 = pack2(*(const uint4*)(ap1 + k),
                                 *(const uint4*)(ap1 + k + 16));
                v16bf g = pack2(*(const uint4*)(bg + k),
                                *(const uint4*)(bg + k + 16));
                v16bf u = pack2(*(const uint4*)(bu + k),
                                *(const uint4*)(bu + k + 16));
                __builtin_prefetch(bg + k + 512, 0, 1);   // global_prefetch_b8
                __builtin_prefetch(bu + k + 512, 0, 1);
                ag0 = __builtin_amdgcn_wmma_f32_16x16x32_bf16(
                          false, a0, false, g, (short)0, ag0, false, false);
                ag1 = __builtin_amdgcn_wmma_f32_16x16x32_bf16(
                          false, a1, false, g, (short)0, ag1, false, false);
                au0 = __builtin_amdgcn_wmma_f32_16x16x32_bf16(
                          false, a0, false, u, (short)0, au0, false, false);
                au1 = __builtin_amdgcn_wmma_f32_16x16x32_bf16(
                          false, a1, false, u, (short)0, au1, false, false);
            }
#pragma unroll
            for (int v = 0; v < 8; ++v) {
                int m = v + 8 * hgrp;
                float h0 = silu_fast(ag0[v]) * au0[v] * wtile[m];
                float h1 = silu_fast(ag1[v]) * au1[v] * wtile[m + 16];
                hs[m * SH + n0 + r15]        = f2bf(h0);
                hs[(m + 16) * SH + n0 + r15] = f2bf(h1);
            }
        }
        __syncthreads();

        // ----- Phase B: out += hidden Wd^T -----
        for (int nc = 0; nc < D_DIM; nc += 128) {
            const int n0 = nc + wave * 16;
            v8f acc0 = {0.f,0.f,0.f,0.f,0.f,0.f,0.f,0.f};
            v8f acc1 = {0.f,0.f,0.f,0.f,0.f,0.f,0.f,0.f};
            const unsigned short* ap0 = hs + r15 * SH + hgrp * 8;
            const unsigned short* ap1 = hs + (r15 + 16) * SH + hgrp * 8;
            const unsigned short* bd = Wd_e + (size_t)(n0 + r15) * H_DIM + hgrp * 8;
#pragma unroll 2
            for (int k = 0; k < H_DIM; k += 32) {
                v16bf a0 = pack2(*(const uint4*)(ap0 + k),
                                 *(const uint4*)(ap0 + k + 16));
                v16bf a1 = pack2(*(const uint4*)(ap1 + k),
                                 *(const uint4*)(ap1 + k + 16));
                v16bf b = pack2(*(const uint4*)(bd + k),
                                *(const uint4*)(bd + k + 16));
                __builtin_prefetch(bd + k + 512, 0, 1);
                acc0 = __builtin_amdgcn_wmma_f32_16x16x32_bf16(
                           false, a0, false, b, (short)0, acc0, false, false);
                acc1 = __builtin_amdgcn_wmma_f32_16x16x32_bf16(
                           false, a1, false, b, (short)0, acc1, false, false);
            }
#pragma unroll
            for (int v = 0; v < 8; ++v) {
                int m = v + 8 * hgrp;
                size_t i0 = (size_t)(tok0 + m) * D_DIM + n0 + r15;
                size_t i1 = (size_t)(tok0 + m + 16) * D_DIM + n0 + r15;
                if (ee == 0) { out[i0] = acc0[v];  out[i1] = acc1[v]; }
                else         { out[i0] += acc0[v]; out[i1] += acc1[v]; }
            }
        }
        __syncthreads();   // hs reused by next expert's Phase A
    }
}

// ---------------------------------------------------------------------------
// Launcher. Workspace layout (~190 MB, assumed <= ws_size):
//   xb, Wgb, Wub, Wdb, Wsg, Wsu, Wsd (bf16 bits), cw (f32).
// ---------------------------------------------------------------------------
extern "C" void kernel_launch(void* const* d_in, const int* in_sizes, int n_in,
                              void* d_out, int out_size, void* d_ws, size_t ws_size,
                              hipStream_t stream) {
    (void)in_sizes; (void)n_in; (void)out_size; (void)ws_size;

    const float* x    = (const float*)d_in[0];
    const float* gw   = (const float*)d_in[1];
    const float* bias = (const float*)d_in[2];
    const float* Wg   = (const float*)d_in[3];
    const float* Wu   = (const float*)d_in[4];
    const float* Wd   = (const float*)d_in[5];
    const float* Wsg  = (const float*)d_in[6];
    const float* Wsu  = (const float*)d_in[7];
    const float* Wsd  = (const float*)d_in[8];
    float* out = (float*)d_out;

    char* ws = (char*)d_ws;
    auto carve = [&](size_t bytes) {
        char* p = ws;
        ws += (bytes + 255) & ~(size_t)255;
        return p;
    };
    const size_t nXD  = (size_t)N_TOK * D_DIM;
    const size_t nEHD = (size_t)E_NUM * H_DIM * D_DIM;
    const size_t nHD  = (size_t)H_DIM * D_DIM;

    unsigned short* xb   = (unsigned short*)carve(nXD  * 2);
    unsigned short* Wgb  = (unsigned short*)carve(nEHD * 2);
    unsigned short* Wub  = (unsigned short*)carve(nEHD * 2);
    unsigned short* Wdb  = (unsigned short*)carve(nEHD * 2);
    unsigned short* Wsgb = (unsigned short*)carve(nHD * 2);
    unsigned short* Wsub = (unsigned short*)carve(nHD * 2);
    unsigned short* Wsdb = (unsigned short*)carve(nHD * 2);
    float* cw = (float*)carve((size_t)N_TOK * E_NUM * 4);

    // Stage 1: bf16 conversion
    cvt_bf16_kernel<<<2048, 256, 0, stream>>>(x,   xb,  (int)nXD);
    cvt_bf16_kernel<<<2048, 256, 0, stream>>>(Wg,  Wgb, (int)nEHD);
    cvt_bf16_kernel<<<2048, 256, 0, stream>>>(Wu,  Wub, (int)nEHD);
    cvt_bf16_kernel<<<2048, 256, 0, stream>>>(Wd,  Wdb, (int)nEHD);
    cvt_bf16_kernel<<<2048, 256, 0, stream>>>(Wsg, Wsgb, (int)nHD);
    cvt_bf16_kernel<<<2048, 256, 0, stream>>>(Wsu, Wsub, (int)nHD);
    cvt_bf16_kernel<<<2048, 256, 0, stream>>>(Wsd, Wsdb, (int)nHD);

    // Stage 2: gating
    gate_kernel<<<N_TOK / 8, 256, 0, stream>>>(x, gw, bias, cw);

    // Stage 3: fused MoE (256 tiles x 256 threads)
    moe_kernel<<<N_TOK / TM, 256, 0, stream>>>(xb, Wgb, Wub, Wdb,
                                               Wsgb, Wsub, Wsdb, cw, out);
}
